// MultiScaleGraphConvolution_26310969655903
// MI455X (gfx1250) — matrix operand order repaired and checked by reading, not compiled, not run on hardware
//
#include <hip/hip_runtime.h>
#include <hip/hip_bf16.h>

typedef __attribute__((ext_vector_type(8)))  _Float16     v8h;
typedef __attribute__((ext_vector_type(16))) _Float16     v16h;
typedef __attribute__((ext_vector_type(8)))  float        v8f;
typedef __attribute__((ext_vector_type(4)))  unsigned int u32x4;
typedef __attribute__((ext_vector_type(8)))  int          i32x8;
typedef __attribute__((ext_vector_type(4)))  int          i32x4;

#define BB    8
#define NNODE 1024
#define DIN   256
#define DOUT  384
#define DHID  128
#define NEDGE 16384
#define ROWS  (BB*NNODE)   // 8192

// ---------------- fragment load helpers (work for global or LDS pointers) ----
// A fragment (row-major): lane holds M=(lane&15); K runs {8h..8h+7},{16+8h..23+8h}
__device__ __forceinline__ v16h ld_afrag(const _Float16* p, int k0, int half) {
  v8h lo = *(const v8h*)(p + k0 + 8*half);
  v8h hi = *(const v8h*)(p + k0 + 16 + 8*half);
  return __builtin_shufflevector(lo, hi, 0,1,2,3,4,5,6,7,8,9,10,11,12,13,14,15);
}
// B fragment from BT (row = output column): K runs {8h..8h+7},{16+8h..23+8h}
__device__ __forceinline__ v16h ld_bfrag(const _Float16* p, int k0, int half) {
  v8h lo = *(const v8h*)(p + k0 + 8*half);
  v8h hi = *(const v8h*)(p + k0 + 16 + 8*half);
  return __builtin_shufflevector(lo, hi, 0,1,2,3,4,5,6,7,8,9,10,11,12,13,14,15);
}
__device__ __forceinline__ v8f wmma(v16h a, v16h b, v8f c) {
  return __builtin_amdgcn_wmma_f32_16x16x32_f16(false, a, false, b, (short)0, c, false, false);
}

// ---------------- Tensor Data Mover: 2D f16 tile -> LDS ----------------------
// Builds D# groups per CDNA5 ISA 8.3/8.4 (count=1, type=2 image, data_size=2B).
__device__ __forceinline__ void tdm_load_2d_f16(unsigned lds_byte_addr, const void* gsrc,
                                                unsigned tensor_d0, unsigned tensor_d1,
                                                unsigned tile_d0, unsigned tile_d1,
                                                unsigned long long stride0_elems) {
  unsigned long long ga = (unsigned long long)gsrc;
  u32x4 g0;
  g0[0] = 1u;                                        // count=1, user-mode D#
  g0[1] = lds_byte_addr;                             // lds_addr
  g0[2] = (unsigned)(ga & 0xffffffffu);              // global_addr[31:0]
  g0[3] = (unsigned)((ga >> 32) & 0x01ffffffu)       // global_addr[56:32]
        | (2u << 30);                                // type = 2
  i32x8 g1;
  g1[0] = (int)(1u << 16);                           // data_size=1 (2 bytes), no flags
  g1[1] = (int)((tensor_d0 & 0xffffu) << 16);        // tensor_dim0[15:0]
  g1[2] = (int)((tensor_d0 >> 16) | ((tensor_d1 & 0xffffu) << 16));
  g1[3] = (int)((tensor_d1 >> 16) | (tile_d0 << 16));// tensor_dim1[31:16] | tile_dim0
  g1[4] = (int)(tile_d1 & 0xffffu);                  // tile_dim1 (tile_dim2 = 0)
  g1[5] = (int)(unsigned)(stride0_elems & 0xffffffffu);     // tensor_dim0_stride[31:0]
  g1[6] = (int)(unsigned)((stride0_elems >> 32) & 0xffffu); // stride[47:32]
  g1[7] = 0;
  i32x4 z4 = {0, 0, 0, 0};
#if __clang_major__ >= 23
  i32x8 z8 = {0, 0, 0, 0, 0, 0, 0, 0};
  __builtin_amdgcn_tensor_load_to_lds(g0, g1, z4, z4, z8, 0);
#else
  __builtin_amdgcn_tensor_load_to_lds(g0, g1, z4, z4, 0);
#endif
}

// ---------------- utility kernels ----------------
__global__ void k_zero_f16(_Float16* p, int n) {
  int i = blockIdx.x*blockDim.x + threadIdx.x;
  if (i < n) p[i] = (_Float16)0.f;
}

__global__ void k_cvt_f16(const float* __restrict__ s, _Float16* __restrict__ d, int n) {
  int i = blockIdx.x*blockDim.x + threadIdx.x;
  if (i < n) d[i] = (_Float16)s[i];
}

// src [R,C] row-major f32 -> dst [C,R] row-major f16 (transposed)
__global__ void k_transpose_cvt(const float* __restrict__ s, _Float16* __restrict__ d, int R, int C) {
  int i = blockIdx.x*blockDim.x + threadIdx.x;
  if (i < R*C) { int r = i / C, c = i % C; d[c*R + r] = (_Float16)s[i]; }
}

__global__ void k_scatter_edges(const int* __restrict__ e,
                                _Float16* __restrict__ A1, _Float16* __restrict__ A1T) {
  int i = blockIdx.x*blockDim.x + threadIdx.x;
  if (i < NEDGE) {
    int s = e[i], d = e[NEDGE + i];
    A1 [s*NNODE + d] = (_Float16)1.f;   // duplicates write same value: benign
    A1T[d*NNODE + s] = (_Float16)1.f;
  }
}

// ---------------- direct-load WMMA GEMM, 32x32 tile per wave ----------------
// D[M,Nc] = A[M,K] * BT[Nc,K]^T
// MODE 0: f32 store (+batched strides)           -> aggregation
// MODE 1: (acc+bias) -> f16 stored TRANSPOSED    -> h (feeds next GEMM's BT)
// MODE 3: exact GELU(acc+bias) -> f32            -> final output
template<int MODE>
__global__ __launch_bounds__(256)
void k_gemm32(const _Float16* __restrict__ A, const _Float16* __restrict__ BT,
              float* __restrict__ Cf, _Float16* __restrict__ Ch,
              const float* __restrict__ bias,
              int M, int Nc, int K, int lda, int ldbt, int ldc,
              long strideA, long strideBT, long strideC)
{
  const int wave = threadIdx.x >> 5;
  const int lane = threadIdx.x & 31;
  const int half = lane >> 4;
  const int n16  = lane & 15;
  const int tilesN = Nc >> 5;
  const int tile = blockIdx.x * (blockDim.x >> 5) + wave;
  if (tile >= (M >> 5) * tilesN) return;            // wave-uniform
  const int row0 = (tile / tilesN) << 5;
  const int col0 = (tile % tilesN) << 5;

  const long zb = blockIdx.z;
  const _Float16* ap0 = A  + zb*strideA  + (long)(row0 + n16) * lda;
  const _Float16* ap1 = ap0 + (long)16 * lda;
  const _Float16* bp0 = BT + zb*strideBT + (long)(col0 + n16) * ldbt;
  const _Float16* bp1 = bp0 + (long)16 * ldbt;

  v8f acc00 = {}, acc01 = {}, acc10 = {}, acc11 = {};
  for (int k0 = 0; k0 < K; k0 += 32) {
    if (k0 + 64 < K) {                              // keep-near prefetch (L2-resident set)
      __builtin_prefetch(ap0 + k0 + 64, 0, 3);
      __builtin_prefetch(ap1 + k0 + 64, 0, 3);
      __builtin_prefetch(bp0 + k0 + 64, 0, 3);
      __builtin_prefetch(bp1 + k0 + 64, 0, 3);
    }
    v16h a0 = ld_afrag(ap0, k0, half);
    v16h a1 = ld_afrag(ap1, k0, half);
    v16h b0 = ld_bfrag(bp0, k0, half);
    v16h b1 = ld_bfrag(bp1, k0, half);
    acc00 = wmma(a0, b0, acc00);
    acc01 = wmma(a0, b1, acc01);
    acc10 = wmma(a1, b0, acc10);
    acc11 = wmma(a1, b1, acc11);
  }

  float* outz = (MODE == 0) ? (Cf + zb*strideC) : Cf;
  auto store_tile = [&](v8f acc, int i, int j) {
    const int rbase = row0 + i*16 + 8*half;          // C layout: VGPR r -> row rbase+r
    const int col   = col0 + j*16 + n16;
    if (MODE == 0) {
      #pragma unroll
      for (int r = 0; r < 8; ++r) outz[(long)(rbase + r)*ldc + col] = acc[r];
    } else if (MODE == 1) {
      float bv = bias[col];
      v8h o;
      #pragma unroll
      for (int r = 0; r < 8; ++r) o[r] = (_Float16)(acc[r] + bv);
      *(v8h*)(Ch + (long)col*ldc + rbase) = o;       // ldc = transposed leading dim (=M)
    } else {
      float bv = bias[col];
      #pragma unroll
      for (int r = 0; r < 8; ++r) {
        float xv = acc[r] + bv;
        outz[(long)(rbase + r)*ldc + col] = 0.5f * xv * (1.f + erff(xv * 0.70710678118654752f));
      }
    }
  };
  store_tile(acc00, 0, 0); store_tile(acc01, 0, 1);
  store_tile(acc10, 1, 0); store_tile(acc11, 1, 1);
}

// ---------------- TDM + LDS staged GEMM for adjacency powers -----------------
// C[1024,1024] = (A[1024,1024] @ BT[1024,1024]^T) > 0, f16 out.
// Block = 256 threads = 8 waves as 2(row)x4(col) of 32x32 tiles -> 64x128 block tile.
// Per K-step the TDM stages A-tile 64x32 and BT-tile 128x32 into LDS (one wave
// issues tensor_load_to_lds; everyone syncs), waves consume via ds_load_b128.
__global__ __launch_bounds__(256)
void k_pow_tdm(const _Float16* __restrict__ A, const _Float16* __restrict__ BT,
               _Float16* __restrict__ Ch)
{
  __shared__ _Float16 sA[64 * 32];    // 4 KB
  __shared__ _Float16 sB[128 * 32];   // 8 KB
  const int wave = threadIdx.x >> 5;
  const int lane = threadIdx.x & 31;
  const int half = lane >> 4;
  const int n16  = lane & 15;
  const int wr = wave >> 2;           // 0..1  (row group)
  const int wc = wave & 3;            // 0..3  (col group)
  const int row0 = blockIdx.y * 64;
  const int col0 = blockIdx.x * 128;

  const unsigned ldsA = (unsigned)(unsigned long long)(void*)sA;  // LDS byte offset
  const unsigned ldsB = (unsigned)(unsigned long long)(void*)sB;

  const _Float16* sa0 = sA + (wr*32      + n16) * 32;
  const _Float16* sa1 = sA + (wr*32 + 16 + n16) * 32;
  const _Float16* sb0 = sB + (wc*32      + n16) * 32;
  const _Float16* sb1 = sB + (wc*32 + 16 + n16) * 32;

  v8f acc00 = {}, acc01 = {}, acc10 = {}, acc11 = {};
  for (int k0 = 0; k0 < NNODE; k0 += 32) {
    __syncthreads();                                 // previous tile fully consumed
    if (wave == 0) {
      tdm_load_2d_f16(ldsA, A  + (unsigned long long)row0*NNODE + k0,
                      NNODE, NNODE, 32, 64, NNODE);
      tdm_load_2d_f16(ldsB, BT + (unsigned long long)col0*NNODE + k0,
                      NNODE, NNODE, 32, 128, NNODE);
      __builtin_amdgcn_s_wait_tensorcnt((short)0);
    }
    __syncthreads();                                 // staged tiles visible to all waves
    v16h a0 = ld_afrag(sa0, 0, half);
    v16h a1 = ld_afrag(sa1, 0, half);
    v16h b0 = ld_bfrag(sb0, 0, half);
    v16h b1 = ld_bfrag(sb1, 0, half);
    acc00 = wmma(a0, b0, acc00);
    acc01 = wmma(a0, b1, acc01);
    acc10 = wmma(a1, b0, acc10);
    acc11 = wmma(a1, b1, acc11);
  }

  auto store_tile = [&](v8f acc, int i, int j) {
    const int rbase = row0 + wr*32 + i*16 + 8*half;
    const int col   = col0 + wc*32 + j*16 + n16;
    #pragma unroll
    for (int r = 0; r < 8; ++r)
      Ch[(long)(rbase + r)*NNODE + col] = (_Float16)(acc[r] > 0.f ? 1.f : 0.f);
  };
  store_tile(acc00, 0, 0); store_tile(acc01, 0, 1);
  store_tile(acc10, 1, 0); store_tile(acc11, 1, 1);
}

// ---------------- BatchNorm (train mode, biased var, eps 1e-5) ----------------
__global__ void k_bn_stats(const float* __restrict__ agg, float* __restrict__ mean,
                           float* __restrict__ inv) {
  const int c = blockIdx.x;                          // one block per channel (384)
  float s = 0.f, s2 = 0.f;
  for (int r = threadIdx.x; r < ROWS; r += blockDim.x) {
    float v = agg[(long)r*DOUT + c]; s += v; s2 += v*v;
  }
  __shared__ float sh[256], sh2[256];
  sh[threadIdx.x] = s; sh2[threadIdx.x] = s2;
  __syncthreads();
  for (int off = 128; off > 0; off >>= 1) {
    if (threadIdx.x < off) { sh[threadIdx.x] += sh[threadIdx.x+off];
                             sh2[threadIdx.x] += sh2[threadIdx.x+off]; }
    __syncthreads();
  }
  if (threadIdx.x == 0) {
    float m   = sh[0]  / (float)ROWS;
    float var = sh2[0] / (float)ROWS - m*m;
    mean[c] = m;
    inv[c]  = rsqrtf(var + 1e-5f);
  }
}

__global__ void k_bn_apply(const float* __restrict__ agg, const float* __restrict__ mean,
                           const float* __restrict__ inv,
                           const float* g0, const float* g1, const float* g2,
                           const float* e0, const float* e1, const float* e2,
                           _Float16* __restrict__ out) {
  int i = blockIdx.x*blockDim.x + threadIdx.x;
  if (i < ROWS*DOUT) {
    int c = i % DOUT;
    int s = c >> 7, j = c & 127;
    const float* g = (s == 0) ? g0 : ((s == 1) ? g1 : g2);
    const float* e = (s == 0) ? e0 : ((s == 1) ? e1 : e2);
    out[i] = (_Float16)((agg[i] - mean[c]) * inv[c] * g[j] + e[j]);
  }
}

// ---------------- host driver ----------------
extern "C" void kernel_launch(void* const* d_in, const int* in_sizes, int n_in,
                              void* d_out, int out_size, void* d_ws, size_t ws_size,
                              hipStream_t stream) {
  const float* x    = (const float*)d_in[0];
  const int*   edge = (const int*)  d_in[1];
  const float* W[3]  = {(const float*)d_in[2],  (const float*)d_in[6],  (const float*)d_in[10]};
  const float* bi[3] = {(const float*)d_in[3],  (const float*)d_in[7],  (const float*)d_in[11]};
  const float* ga[3] = {(const float*)d_in[4],  (const float*)d_in[8],  (const float*)d_in[12]};
  const float* be[3] = {(const float*)d_in[5],  (const float*)d_in[9],  (const float*)d_in[13]};
  const float* Wfm  = (const float*)d_in[14];
  const float* bfv  = (const float*)d_in[15];
  float* out = (float*)d_out;

  // bump allocator over workspace
  char* ws = (char*)d_ws;
  auto alloc = [&](size_t bytes) -> char* {
    char* p = ws; ws += (bytes + 255) & ~(size_t)255; return p;
  };
  _Float16* A1   = (_Float16*)alloc((size_t)NNODE*NNODE*2);  // adjacency (f16, 0/1)
  _Float16* A1T  = (_Float16*)alloc((size_t)NNODE*NNODE*2);  // adjacency transposed
  _Float16* A2   = (_Float16*)alloc((size_t)NNODE*NNODE*2);  // (A@A > 0)
  _Float16* A3   = (_Float16*)alloc((size_t)NNODE*NNODE*2);  // (A2@A > 0)
  _Float16* x16  = (_Float16*)alloc((size_t)ROWS*DIN*2);     // x as f16
  _Float16* WT   = (_Float16*)alloc((size_t)3*DHID*DIN*2);   // W_i^T f16
  _Float16* WfT  = (_Float16*)alloc((size_t)DOUT*DOUT*2);    // Wf^T f16
  _Float16* hT   = (_Float16*)alloc((size_t)3*DHID*ROWS*2);  // h_i^T [128 x 8192] f16
  float*    agg  = (float*)   alloc((size_t)ROWS*DOUT*4);    // pre-BN concat, f32
  _Float16* cc16 = (_Float16*)alloc((size_t)ROWS*DOUT*2);    // normalized concat f16
  float*    meanb= (float*)   alloc((size_t)DOUT*4);
  float*    invb = (float*)   alloc((size_t)DOUT*4);

  const dim3 blk(256);

  // 1) clear adjacency (+transpose, contiguous), build from edges
  k_zero_f16<<<dim3((2*NNODE*NNODE + 255)/256), blk, 0, stream>>>(A1, 2*NNODE*NNODE);
  k_scatter_edges<<<dim3((NEDGE + 255)/256), blk, 0, stream>>>(edge, A1, A1T);

  // 2) f16 conversions (x plain; weights transposed for the BT operand)
  k_cvt_f16<<<dim3((ROWS*DIN + 255)/256), blk, 0, stream>>>(x, x16, ROWS*DIN);
  for (int i = 0; i < 3; ++i)
    k_transpose_cvt<<<dim3((DIN*DHID + 255)/256), blk, 0, stream>>>(
        W[i], WT + (size_t)i*DHID*DIN, DIN, DHID);
  k_transpose_cvt<<<dim3((DOUT*DOUT + 255)/256), blk, 0, stream>>>(Wfm, WfT, DOUT, DOUT);

  // 3) h_i = x @ W_i + b_i, stored transposed f16  (M=8192, Nc=128, K=256)
  //    tiles = 256*4 = 1024 -> 128 blocks of 8 waves
  for (int i = 0; i < 3; ++i)
    k_gemm32<1><<<dim3(128), blk, 0, stream>>>(
        x16, WT + (size_t)i*DHID*DIN, nullptr, hT + (size_t)i*DHID*ROWS, bi[i],
        ROWS, DHID, DIN, DIN, DIN, ROWS, 0, 0, 0);

  // 4) adjacency powers via TDM-staged LDS GEMM (exact in f16/f32-acc):
  //    A2 = (A1@A1 > 0), A3 = (A2@A1 > 0); grid = 1024/128 x 1024/64
  k_pow_tdm<<<dim3(NNODE/128, NNODE/64), blk, 0, stream>>>(A1, A1T, A2);
  k_pow_tdm<<<dim3(NNODE/128, NNODE/64), blk, 0, stream>>>(A2, A1T, A3);

  // 5) aggregation agg[b,:,i*128:...] = A_i @ h_i[b]  (M=1024, Nc=128, K=1024, z=batch)
  //    tiles = 32*4 = 128 -> 16 blocks, z = 8
  const _Float16* Apow[3] = {A1, A2, A3};
  for (int i = 0; i < 3; ++i)
    k_gemm32<0><<<dim3(16, 1, BB), blk, 0, stream>>>(
        Apow[i], hT + (size_t)i*DHID*ROWS, agg + (size_t)i*DHID, nullptr, nullptr,
        NNODE, DHID, NNODE, NNODE, ROWS, DOUT,
        /*strideA=*/0, /*strideBT=*/NNODE, /*strideC=*/(long)NNODE*DOUT);

  // 6) BatchNorm over (B*N, 384), normalized concat -> f16
  k_bn_stats<<<dim3(DOUT), blk, 0, stream>>>(agg, meanb, invb);
  k_bn_apply<<<dim3((ROWS*DOUT + 255)/256), blk, 0, stream>>>(
      agg, meanb, invb, ga[0], ga[1], ga[2], be[0], be[1], be[2], cc16);

  // 7) final: GELU(concat @ Wf + bf)  (M=8192, Nc=384, K=384)
  //    tiles = 256*12 = 3072 -> 384 blocks
  k_gemm32<3><<<dim3(384), blk, 0, stream>>>(
      cc16, WfT, out, nullptr, bfv, ROWS, DOUT, DOUT, DOUT, DOUT, DOUT, 0, 0, 0);

  (void)in_sizes; (void)n_in; (void)out_size; (void)ws_size;
}
